// MultiHeadAttention_26414048870884
// MI455X (gfx1250) — matrix operand order repaired and checked by reading, not compiled, not run on hardware
//
#include <hip/hip_runtime.h>

// ---------------- constants ----------------
constexpr int B_ = 8, T_ = 1024, C_ = 768, H_ = 12, DH_ = 64;

// ---------------- types ----------------
typedef __attribute__((ext_vector_type(16))) __bf16 bf16x16;
typedef __attribute__((ext_vector_type(8)))  __bf16 bf16x8;
typedef __attribute__((ext_vector_type(2)))  __bf16 bf16x2;
typedef __attribute__((ext_vector_type(8)))  float  f32x8;

union FragA { bf16x16 v; bf16x8 h[2]; };

__device__ __forceinline__ bf16x2 pk2(float a, float b) {
    bf16x2 r; r.x = (__bf16)a; r.y = (__bf16)b; return r;
}

// =====================================================================
// Kernel 1: QKV projection.  x[b] (1024x768) @ W[h] (768x64) -> bf16.
// grid = (T/128, B*H, 3);  block = 256 (8 waves, 16 rows each).
// z=0 -> Q[B,H,T,DH], z=1 -> K[B,H,T,DH], z=2 -> V^T [B,H,DH,T]
// =====================================================================
__global__ __launch_bounds__(256) void qkv_kernel(
    const float* __restrict__ x, const float* __restrict__ Wq,
    const float* __restrict__ Wk, const float* __restrict__ Wv,
    __bf16* __restrict__ q_ws, __bf16* __restrict__ k_ws,
    __bf16* __restrict__ vt_ws)
{
    __shared__ __align__(16) __bf16 xs[128 * 40];   // 128 rows x 32 k (pad->40)
    __shared__ __align__(16) __bf16 wt[64 * 40];    // 64 n x 32 k (transposed)

    const int tid  = threadIdx.x;
    const int lane = tid & 31;
    const int wave = tid >> 5;
    const int half = lane >> 4;
    const int l15  = lane & 15;

    const int rt = blockIdx.x;             // row tile of 128
    const int bh = blockIdx.y;             // b*H + h
    const int b  = bh / H_;
    const int h  = bh - b * H_;
    const int z  = blockIdx.z;             // 0=Q 1=K 2=V

    const float* W     = (z == 0) ? Wq : (z == 1) ? Wk : Wv;
    const float* Whead = W + (size_t)h * C_ * DH_;
    const float* xb    = x + ((size_t)b * T_ + (size_t)rt * 128) * C_;

    f32x8 acc[4] = {};

    for (int ck = 0; ck < C_; ck += 32) {
        // ---- stage x tile (f32 -> bf16, packed pair stores), 2 threads/row ----
        {
            const int row  = tid >> 1;
            const int hcol = (tid & 1) * 16;
            const float4* src = (const float4*)(xb + (size_t)row * C_ + ck + hcol);
            __bf16* dst = &xs[row * 40 + hcol];
            #pragma unroll
            for (int j = 0; j < 4; ++j) {
                float4 v = src[j];
                *(bf16x2*)(dst + 4 * j)     = pk2(v.x, v.y);
                *(bf16x2*)(dst + 4 * j + 2) = pk2(v.z, v.w);
            }
        }
        // ---- stage W chunk transposed: wt[n][k] ----
        {
            const int k  = tid >> 3;             // 0..31
            const int ng = (tid & 7) * 8;        // n group
            const float4* src = (const float4*)(Whead + (size_t)(ck + k) * DH_ + ng);
            float4 v0 = src[0], v1 = src[1];
            wt[(ng + 0) * 40 + k] = (__bf16)v0.x; wt[(ng + 1) * 40 + k] = (__bf16)v0.y;
            wt[(ng + 2) * 40 + k] = (__bf16)v0.z; wt[(ng + 3) * 40 + k] = (__bf16)v0.w;
            wt[(ng + 4) * 40 + k] = (__bf16)v1.x; wt[(ng + 5) * 40 + k] = (__bf16)v1.y;
            wt[(ng + 6) * 40 + k] = (__bf16)v1.z; wt[(ng + 7) * 40 + k] = (__bf16)v1.w;
        }
        __syncthreads();

        // ---- A fragment (16x32 bf16, per documented lane layout) ----
        FragA a;
        {
            const __bf16* p = &xs[(wave * 16 + l15) * 40 + half * 8];
            a.h[0] = *(const bf16x8*)(p);
            a.h[1] = *(const bf16x8*)(p + 16);
        }
        #pragma unroll
        for (int nt = 0; nt < 4; ++nt) {
            FragA bf;
            const __bf16* p = &wt[(nt * 16 + l15) * 40 + half * 8];
            bf.h[0] = *(const bf16x8*)(p);
            bf.h[1] = *(const bf16x8*)(p + 16);
            acc[nt] = __builtin_amdgcn_wmma_f32_16x16x32_bf16(
                false, a.v, false, bf.v, (short)0, acc[nt], false, false);
        }
        __syncthreads();
    }

    // ---- epilogue ----
    const int row0 = rt * 128 + wave * 16 + half * 8;
    if (z < 2) {
        __bf16* outp = (z == 0 ? q_ws : k_ws) + (size_t)bh * T_ * DH_;
        #pragma unroll
        for (int nt = 0; nt < 4; ++nt) {
            const int col = nt * 16 + l15;
            #pragma unroll
            for (int i = 0; i < 8; ++i)
                outp[(size_t)(row0 + i) * DH_ + col] = (__bf16)acc[nt][i];
        }
    } else {
        __bf16* outp = vt_ws + (size_t)bh * DH_ * T_;   // V^T: [DH][T]
        #pragma unroll
        for (int nt = 0; nt < 4; ++nt) {
            const int d = nt * 16 + l15;
            #pragma unroll
            for (int i = 0; i < 8; ++i)
                outp[(size_t)d * T_ + row0 + i] = (__bf16)acc[nt][i];
        }
    }
}

// =====================================================================
// Kernel 2: causal flash attention per (b,h).
// grid = (T/64, B*H); block = 128 (4 waves, 16 query rows each).
// =====================================================================
__global__ __launch_bounds__(128) void attn_kernel(
    const __bf16* __restrict__ q_ws,
    const __bf16* __restrict__ k_ws,
    const __bf16* __restrict__ vt_ws,
    __bf16* __restrict__ attn_ws)
{
    __shared__ __align__(16) __bf16 plds[4][16 * 32];  // per-wave P tile

    const int tid  = threadIdx.x;
    const int lane = tid & 31;
    const int wave = tid >> 5;
    const int half = lane >> 4;
    const int l15  = lane & 15;

    const int bh = blockIdx.y;
    const int b  = bh / H_;
    const int h  = bh - b * H_;
    const int qw = blockIdx.x * 64 + wave * 16;   // query-row base of this wave

    const __bf16* qb = q_ws  + (size_t)bh * T_ * DH_;
    const __bf16* kb = k_ws  + (size_t)bh * T_ * DH_;
    const __bf16* vb = vt_ws + (size_t)bh * DH_ * T_;

    // Q fragments (two 16x32 d-chunks), kept in registers for whole kernel
    FragA qf[2];
    #pragma unroll
    for (int dc = 0; dc < 2; ++dc) {
        const __bf16* p = qb + (size_t)(qw + l15) * DH_ + dc * 32 + half * 8;
        qf[dc].h[0] = *(const bf16x8*)(p);
        qf[dc].h[1] = *(const bf16x8*)(p + 16);
    }

    f32x8 o[4] = {};
    float m[8], l[8];
    #pragma unroll
    for (int i = 0; i < 8; ++i) { m[i] = -1e30f; l[i] = 0.0f; }

    const float scale = 0.125f;                    // 1/sqrt(DH)
    const int   nk    = qw + 16;                   // keys needed (causal)
    __bf16* pbuf = plds[wave];

    for (int k0 = 0; k0 < nk; k0 += 32) {
        // prefetch next key/value step (L2-resident; hide latency behind WMMA)
        if (k0 + 32 < nk) {
            __builtin_prefetch(kb + (size_t)(k0 + 32 + l15) * DH_, 0, 1);
            __builtin_prefetch(vb + (size_t)(l15) * T_ + k0 + 32, 0, 1);
        }

        // ---- S = scale * Q K^T for two 16-key tiles ----
        f32x8 s0 = {}, s1 = {};
        #pragma unroll
        for (int dc = 0; dc < 2; ++dc) {
            FragA kf;
            const __bf16* p0 = kb + (size_t)(k0 + l15) * DH_ + dc * 32 + half * 8;
            kf.h[0] = *(const bf16x8*)(p0);
            kf.h[1] = *(const bf16x8*)(p0 + 16);
            s0 = __builtin_amdgcn_wmma_f32_16x16x32_bf16(
                false, qf[dc].v, false, kf.v, (short)0, s0, false, false);
            const __bf16* p1 = kb + (size_t)(k0 + 16 + l15) * DH_ + dc * 32 + half * 8;
            kf.h[0] = *(const bf16x8*)(p1);
            kf.h[1] = *(const bf16x8*)(p1 + 16);
            s1 = __builtin_amdgcn_wmma_f32_16x16x32_bf16(
                false, qf[dc].v, false, kf.v, (short)0, s1, false, false);
        }

        // ---- online softmax (row stats replicated over half-wave lanes) ----
        const bool need_mask = (k0 + 31 > qw);
        const int  key0 = k0 + l15;
        const int  key1 = k0 + 16 + l15;
        #pragma unroll
        for (int i = 0; i < 8; ++i) {
            const int row = qw + half * 8 + i;
            float a0 = s0[i] * scale;
            float a1 = s1[i] * scale;
            if (need_mask) {
                if (key0 > row) a0 = -3.0e38f;
                if (key1 > row) a1 = -3.0e38f;
            }
            float mx = fmaxf(a0, a1);
            mx = fmaxf(mx, __shfl_xor(mx, 1, 32));
            mx = fmaxf(mx, __shfl_xor(mx, 2, 32));
            mx = fmaxf(mx, __shfl_xor(mx, 4, 32));
            mx = fmaxf(mx, __shfl_xor(mx, 8, 32));
            const float mnew  = fmaxf(m[i], mx);
            const float alpha = __expf(m[i] - mnew);
            const float p0 = __expf(a0 - mnew);
            const float p1 = __expf(a1 - mnew);
            float rs = p0 + p1;
            rs += __shfl_xor(rs, 1, 32);
            rs += __shfl_xor(rs, 2, 32);
            rs += __shfl_xor(rs, 4, 32);
            rs += __shfl_xor(rs, 8, 32);
            l[i] = l[i] * alpha + rs;
            m[i] = mnew;
            #pragma unroll
            for (int nt = 0; nt < 4; ++nt) o[nt][i] *= alpha;
            const int prow = half * 8 + i;
            pbuf[prow * 32 + l15]      = (__bf16)p0;
            pbuf[prow * 32 + 16 + l15] = (__bf16)p1;
        }

        // wave-local DS ops are hardware-ordered; just stop compiler reordering
        asm volatile("" ::: "memory");

        // ---- P (D-layout) -> A-fragment via LDS ----
        FragA pf;
        {
            const __bf16* p = &pbuf[l15 * 32 + half * 8];
            pf.h[0] = *(const bf16x8*)(p);
            pf.h[1] = *(const bf16x8*)(p + 16);
        }
        // ---- O += P @ V  (V^T gives contiguous B-fragment loads) ----
        #pragma unroll
        for (int nt = 0; nt < 4; ++nt) {
            FragA vf;
            const __bf16* p = vb + (size_t)(nt * 16 + l15) * T_ + k0 + half * 8;
            vf.h[0] = *(const bf16x8*)(p);
            vf.h[1] = *(const bf16x8*)(p + 16);
            o[nt] = __builtin_amdgcn_wmma_f32_16x16x32_bf16(
                false, pf.v, false, vf.v, (short)0, o[nt], false, false);
        }
    }

    // ---- normalize + write concat-head layout [B,T,H*DH] as bf16 ----
    __bf16* outp = attn_ws + (size_t)b * T_ * (H_ * DH_) + (size_t)h * DH_;
    #pragma unroll
    for (int i = 0; i < 8; ++i) {
        const float inv = 1.0f / l[i];
        const int   row = qw + half * 8 + i;
        #pragma unroll
        for (int nt = 0; nt < 4; ++nt)
            outp[(size_t)row * (H_ * DH_) + nt * 16 + l15] = (__bf16)(o[nt][i] * inv);
    }
}

// =====================================================================
// Kernel 3: output projection  attn[8192,768](bf16) @ Wp[768,768] + bp -> f32
// grid = (B*T/128, C/64); block = 256 (8 waves).
// =====================================================================
__global__ __launch_bounds__(256) void proj_kernel(
    const __bf16* __restrict__ attn_ws,
    const float* __restrict__ Wp, const float* __restrict__ bp,
    float* __restrict__ out)
{
    __shared__ __align__(16) __bf16 wt[64 * 40];

    const int tid  = threadIdx.x;
    const int lane = tid & 31;
    const int wave = tid >> 5;
    const int half = lane >> 4;
    const int l15  = lane & 15;

    const int rt = blockIdx.x;   // 64 row tiles of 128 over B*T
    const int ct = blockIdx.y;   // 12 col tiles of 64

    const __bf16* abase = attn_ws + (size_t)(rt * 128 + wave * 16 + l15) * C_;

    f32x8 acc[4] = {};

    for (int ck = 0; ck < C_; ck += 32) {
        // stage Wp chunk transposed: wt[n][k] = Wp[ck+k][ct*64+n]
        {
            const int k  = tid >> 3;
            const int ng = (tid & 7) * 8;
            const float4* src = (const float4*)(Wp + (size_t)(ck + k) * C_ + ct * 64 + ng);
            float4 v0 = src[0], v1 = src[1];
            wt[(ng + 0) * 40 + k] = (__bf16)v0.x; wt[(ng + 1) * 40 + k] = (__bf16)v0.y;
            wt[(ng + 2) * 40 + k] = (__bf16)v0.z; wt[(ng + 3) * 40 + k] = (__bf16)v0.w;
            wt[(ng + 4) * 40 + k] = (__bf16)v1.x; wt[(ng + 5) * 40 + k] = (__bf16)v1.y;
            wt[(ng + 6) * 40 + k] = (__bf16)v1.z; wt[(ng + 7) * 40 + k] = (__bf16)v1.w;
        }
        __syncthreads();

        FragA a;   // A fragment straight from global (bf16 row-major matches layout)
        {
            const __bf16* p = abase + ck + half * 8;
            a.h[0] = *(const bf16x8*)(p);
            a.h[1] = *(const bf16x8*)(p + 16);
        }
        #pragma unroll
        for (int nt = 0; nt < 4; ++nt) {
            FragA bf;
            const __bf16* p = &wt[(nt * 16 + l15) * 40 + half * 8];
            bf.h[0] = *(const bf16x8*)(p);
            bf.h[1] = *(const bf16x8*)(p + 16);
            acc[nt] = __builtin_amdgcn_wmma_f32_16x16x32_bf16(
                false, a.v, false, bf.v, (short)0, acc[nt], false, false);
        }
        __syncthreads();
    }

    const int row0 = rt * 128 + wave * 16 + half * 8;
    #pragma unroll
    for (int nt = 0; nt < 4; ++nt) {
        const int col  = ct * 64 + nt * 16 + l15;
        const float bb = bp[col];
        #pragma unroll
        for (int i = 0; i < 8; ++i)
            out[(size_t)(row0 + i) * C_ + col] = acc[nt][i] + bb;
    }
}

// =====================================================================
extern "C" void kernel_launch(void* const* d_in, const int* in_sizes, int n_in,
                              void* d_out, int out_size, void* d_ws, size_t ws_size,
                              hipStream_t stream) {
    (void)in_sizes; (void)n_in; (void)out_size; (void)ws_size;

    const float* x  = (const float*)d_in[0];
    const float* Wq = (const float*)d_in[1];
    const float* Wk = (const float*)d_in[2];
    const float* Wv = (const float*)d_in[3];
    const float* Wp = (const float*)d_in[4];
    const float* bp = (const float*)d_in[5];
    float* out = (float*)d_out;

    const size_t seg = (size_t)B_ * H_ * T_ * DH_;   // 6,291,456 bf16 elems
    __bf16* q_ws  = (__bf16*)d_ws;
    __bf16* k_ws  = q_ws  + seg;
    __bf16* vt_ws = k_ws  + seg;
    __bf16* at_ws = vt_ws + seg;                     // total 48 MB of d_ws

    qkv_kernel<<<dim3(T_ / 128, B_ * H_, 3), 256, 0, stream>>>(
        x, Wq, Wk, Wv, q_ws, k_ws, vt_ws);
    attn_kernel<<<dim3(T_ / 64, B_ * H_), 128, 0, stream>>>(
        q_ws, k_ws, vt_ws, at_ws);
    proj_kernel<<<dim3((B_ * T_) / 128, C_ / 64), 256, 0, stream>>>(
        at_ws, Wp, bp, out);
}
